// GNS_24026047054803
// MI455X (gfx1250) — matrix operand order repaired and checked by reading, not compiled, not run on hardware
//
#include <hip/hip_runtime.h>
#include <cmath>

#define NBUS   100000
#define NEDGE  1600000
#define NGEN   5000
#define KITER  10
#define LATENT 10
#define HID    20
#define ALPHA  0.1f
#define SLOPE  0.01f

typedef __attribute__((ext_vector_type(16))) _Float16 v16h;
typedef __attribute__((ext_vector_type(8)))  float    v8f;

union H16 { v16h v; _Float16 h[16]; unsigned int u[8]; };
union F8  { v8f  v; float f[8]; };

struct MlpW { const float *W1,*B1,*W2,*B2,*W4,*B4; };

// ---- WMMA fragment helpers (CDNA5 ISA 16-bit layouts, wave32) -------------

// A-matrix 16x32 f16: lanes<16 hold K 0..7 & 16..23, lanes>=16 hold K 8..15 & 24..31
__device__ inline int kmapA(int lane, int h){
  int g = h >> 1;
  int k = (g < 4) ? (g << 1) : (16 + ((g - 4) << 1));
  return k + (h & 1) + ((lane & 16) ? 8 : 0);
}
// B-matrix 32x16 f16: lanes<16 hold K 0..7 & 16..23, lanes>=16 hold K 8..15 & 24..31 (lane = N)
__device__ inline int kmapB(int lane, int h){
  return h + ((h >= 8) ? 8 : 0) + ((lane & 16) ? 8 : 0);
}

__device__ inline v8f wmma16(v16h a, v16h b, v8f c){
  return __builtin_amdgcn_wmma_f32_16x16x32_f16(false, a, false, b, (short)0, c, false, false);
}

__device__ inline unsigned int pack2(float a, float b){
  union { _Float16 hh[2]; unsigned int uu; } c;
  c.hh[0] = (_Float16)a; c.hh[1] = (_Float16)b;
  return c.uu;
}

// Pack one B-fragment (din x dout slice, ldw = row stride of W, ntile picks N 0..15 / 16..31)
// into LDS as 32 lanes x 8 dwords. Block-cooperative.
__device__ inline void pack_b_lds(unsigned int* dst, const float* W, int din, int dout, int ldw, int ntile){
  for (int idx = threadIdx.x; idx < 256; idx += blockDim.x){
    int lane = idx >> 3, d = idx & 7;
    int n  = (lane & 15) + (ntile << 4);
    int k0 = kmapB(lane, 2*d), k1 = k0 + 1;
    float lo = 0.f, hi = 0.f;
    if (n < dout){
      if (k0 < din) lo = W[k0*ldw + n];
      if (k1 < din) hi = W[k1*ldw + n];
    }
    dst[idx] = pack2(lo, hi);
  }
}

__device__ inline v16h load_bfrag(const unsigned int* src, int lane){
  H16 b;
#pragma unroll
  for (int d = 0; d < 8; ++d) b.u[d] = src[lane*8 + d];
  return b.v;
}

__device__ inline v16h load_afrag(const _Float16* xt, int lane){
  H16 a;
  int M = lane & 15;
#pragma unroll
  for (int d = 0; d < 8; ++d){
    int k = kmapA(lane, 2*d);                       // even -> dword aligned
    a.u[d] = *(const unsigned int*)(xt + M*32 + k);
  }
  return a.v;
}

// C/D f32 16x16: lane = N (+16 split selects M 0..7 / 8..15), vgpr r = M row.
// Apply (preloaded) bias + leaky_relu and write f16 hidden tile (zeros in pad cols).
__device__ inline void store_act(_Float16* xt, int lane, v8f cv, float bv, int dout, int ntile){
  F8 c; c.v = cv;
  int n  = (lane & 15) + (ntile << 4);
  int mb = (lane & 16) ? 8 : 0;
#pragma unroll
  for (int r = 0; r < 8; ++r){
    float val = 0.f;
    if (n < dout){
      float t = c.f[r] + bv;
      val = (t > 0.f) ? t : SLOPE * t;
    }
    xt[(mb + r)*32 + n] = (_Float16)val;
  }
}

// ---- reductions ------------------------------------------------------------

__device__ inline float warp_sum(float v){
#pragma unroll
  for (int o = 16; o > 0; o >>= 1) v += __shfl_down(v, o, 32);
  return v;
}
__device__ inline float block_sum(float v, float* sh){
  int lane = threadIdx.x & 31, wid = threadIdx.x >> 5;
  v = warp_sum(v);
  if (lane == 0) sh[wid] = v;
  __syncthreads();
  float r = 0.f;
  if (wid == 0){
    r = (lane < (int)(blockDim.x >> 5)) ? sh[lane] : 0.f;
    r = warp_sum(r);
  }
  __syncthreads();
  return r;  // valid in thread 0
}

// ---- edge MLP (phi_from): m[dst]||edge_attr -> 20 -> 20 -> 10, scatter to src

__global__ __launch_bounds__(256) void k_edge_mlp(
    const float* __restrict__ lines, const float* __restrict__ m,
    float* __restrict__ phi_sum,
    const float* W1, const float* B1, const float* W2, const float* B2,
    const float* W4, const float* B4)
{
  __shared__ unsigned int bf[5][256];
  __shared__ __align__(16) _Float16 xt[8][512];
  __shared__ int srci[8][16];

  pack_b_lds(bf[0], W1, LATENT+5, HID, HID, 0);
  pack_b_lds(bf[1], W1, LATENT+5, HID, HID, 1);
  pack_b_lds(bf[2], W2, HID, HID, HID, 0);
  pack_b_lds(bf[3], W2, HID, HID, HID, 1);
  pack_b_lds(bf[4], W4, HID, LATENT, LATENT, 0);
  __syncthreads();

  int lane = threadIdx.x & 31, wid = threadIdx.x >> 5;
  int wgid = blockIdx.x * 8 + wid;
  int nwav = gridDim.x * 8;
  const int ntiles = NEDGE / 16;
  int iters = (ntiles + nwav - 1) / nwav;
  _Float16* X = xt[wid];

  // loop-invariant bias values for this lane's output columns
  int n0 = lane & 15, n1 = n0 + 16;
  float b1_0 = B1[n0];
  float b1_1 = (n1 < HID) ? B1[n1] : 0.f;
  float b2_0 = B2[n0];
  float b2_1 = (n1 < HID) ? B2[n1] : 0.f;
  float b4_0 = (n0 < LATENT) ? B4[n0] : 0.f;

  for (int it = 0; it < iters; ++it){
    int t = wgid + it * nwav;
    bool act = (t < ntiles);
    if (act){
      int row = lane & 15;
      int e = t*16 + row;
      unsigned int* Xr = (unsigned int*)(X + row*32);
      if (lane < 16){
        int dd = (int)lines[e*7 + 1] - 1;
        srci[wid][row] = (int)lines[e*7 + 0] - 1;
#pragma unroll
        for (int j = 0; j < LATENT; j += 2)
          Xr[j >> 1] = pack2(m[dd*LATENT + j], m[dd*LATENT + j + 1]);
        // prefetch next tile's line rows (predictable stream)
        int tn = t + nwav;
        if (tn < ntiles)
          __builtin_prefetch(&lines[(size_t)(tn*16 + row)*7], 0, 1);
      } else {
        Xr[5] = pack2(lines[e*7 + 2], lines[e*7 + 3]);
        Xr[6] = pack2(lines[e*7 + 4], lines[e*7 + 5]);
        Xr[7] = pack2(lines[e*7 + 6], 0.f);
#pragma unroll
        for (int j = 8; j < 16; ++j) Xr[j] = 0u;
      }
    }
    __syncthreads();
    if (act){                                    // layer 1
      v16h a = load_afrag(X, lane); v8f z = {};
      v8f c0 = wmma16(a, load_bfrag(bf[0], lane), z);
      v8f c1 = wmma16(a, load_bfrag(bf[1], lane), z);
      store_act(X, lane, c0, b1_0, HID, 0);
      store_act(X, lane, c1, b1_1, HID, 1);
    }
    __syncthreads();
    if (act){                                    // layer 2
      v16h a = load_afrag(X, lane); v8f z = {};
      v8f c0 = wmma16(a, load_bfrag(bf[2], lane), z);
      v8f c1 = wmma16(a, load_bfrag(bf[3], lane), z);
      store_act(X, lane, c0, b2_0, HID, 0);
      store_act(X, lane, c1, b2_1, HID, 1);
    }
    __syncthreads();
    if (act){                                    // layer 3 + scatter
      v16h a = load_afrag(X, lane); v8f z = {};
      F8 c; c.v = wmma16(a, load_bfrag(bf[4], lane), z);
      int mb = (lane & 16) ? 8 : 0;
      if (n0 < LATENT){
#pragma unroll
        for (int r = 0; r < 8; ++r){
          int s = srci[wid][mb + r];
          atomicAdd(&phi_sum[s*LATENT + n0], c.f[r] + b4_0);
        }
      }
    }
    __syncthreads();
  }
}

// ---- bus MLPs (L_theta, L_v, L_m) + state update + sum(v^2) ----------------

__global__ __launch_bounds__(256) void k_bus_mlp(
    float* __restrict__ v, float* __restrict__ theta,
    const float* __restrict__ dp, const float* __restrict__ dq,
    float* __restrict__ m, const float* __restrict__ phi,
    const float* __restrict__ nongen, float* __restrict__ sumv2,
    MlpW wt, MlpW wv, MlpW wm)
{
  __shared__ unsigned int bf[15][256];
  __shared__ __align__(16) _Float16 xt[8][512];
  __shared__ __align__(16) _Float16 ht[8][512];
  __shared__ float shred[8];

  const MlpW* heads[3] = { &wt, &wv, &wm };
  for (int hI = 0; hI < 3; ++hI){
    int d3 = (hI == 2) ? LATENT : 1;
    pack_b_lds(bf[hI*5+0], heads[hI]->W1, 24, HID, HID, 0);
    pack_b_lds(bf[hI*5+1], heads[hI]->W1, 24, HID, HID, 1);
    pack_b_lds(bf[hI*5+2], heads[hI]->W2, HID, HID, HID, 0);
    pack_b_lds(bf[hI*5+3], heads[hI]->W2, HID, HID, HID, 1);
    pack_b_lds(bf[hI*5+4], heads[hI]->W4, HID, d3, d3, 0);
  }
  __syncthreads();

  int lane = threadIdx.x & 31, wid = threadIdx.x >> 5;
  int wgid = blockIdx.x * 8 + wid;
  int nwav = gridDim.x * 8;
  const int ntiles = NBUS / 16;
  int iters = (ntiles + nwav - 1) / nwav;
  _Float16* X  = xt[wid];
  _Float16* Hh = ht[wid];
  float accv2 = 0.f;

  // loop-invariant per-lane biases (per head)
  int n0 = lane & 15, n1 = n0 + 16;
  float hb1[3][2], hb2[3][2];
#pragma unroll
  for (int hI = 0; hI < 3; ++hI){
    hb1[hI][0] = heads[hI]->B1[n0];
    hb1[hI][1] = (n1 < HID) ? heads[hI]->B1[n1] : 0.f;
    hb2[hI][0] = heads[hI]->B2[n0];
    hb2[hI][1] = (n1 < HID) ? heads[hI]->B2[n1] : 0.f;
  }
  float b4t = wt.B4[0];
  float b4v = wv.B4[0];
  float b4m = (n0 < LATENT) ? wm.B4[n0] : 0.f;

  for (int it = 0; it < iters; ++it){
    int t = wgid + it * nwav;
    bool act = (t < ntiles);
    if (act){
      int row = lane & 15;
      int bus = t*16 + row;
      unsigned int* Xr = (unsigned int*)(X + row*32);
      if (lane < 16){
        Xr[0] = pack2(v[bus], theta[bus]);
        Xr[1] = pack2(dp[bus], dq[bus]);
#pragma unroll
        for (int j = 0; j < LATENT; j += 2)
          Xr[2 + (j >> 1)] = pack2(m[bus*LATENT + j], m[bus*LATENT + j + 1]);
      } else {
#pragma unroll
        for (int j = 0; j < LATENT; j += 2)
          Xr[7 + (j >> 1)] = pack2(phi[bus*LATENT + j], phi[bus*LATENT + j + 1]);
#pragma unroll
        for (int j = 12; j < 16; ++j) Xr[j] = 0u;
      }
    }
    __syncthreads();
    for (int hI = 0; hI < 3; ++hI){
      if (act){
        v16h a = load_afrag(X, lane); v8f z = {};
        v8f c0 = wmma16(a, load_bfrag(bf[hI*5+0], lane), z);
        v8f c1 = wmma16(a, load_bfrag(bf[hI*5+1], lane), z);
        store_act(Hh, lane, c0, hb1[hI][0], HID, 0);
        store_act(Hh, lane, c1, hb1[hI][1], HID, 1);
      }
      __syncthreads();
      if (act){
        v16h a = load_afrag(Hh, lane); v8f z = {};
        v8f c0 = wmma16(a, load_bfrag(bf[hI*5+2], lane), z);
        v8f c1 = wmma16(a, load_bfrag(bf[hI*5+3], lane), z);
        store_act(Hh, lane, c0, hb2[hI][0], HID, 0);
        store_act(Hh, lane, c1, hb2[hI][1], HID, 1);
      }
      __syncthreads();
      if (act){
        v16h a = load_afrag(Hh, lane); v8f z = {};
        F8 c; c.v = wmma16(a, load_bfrag(bf[hI*5+4], lane), z);
        int mb = (lane & 16) ? 8 : 0;
        if (hI == 0){
          if (n0 == 0){
#pragma unroll
            for (int r = 0; r < 8; ++r){ int bus = t*16+mb+r; theta[bus] += (c.f[r]+b4t)*ALPHA; }
          }
        } else if (hI == 1){
          if (n0 == 0){
#pragma unroll
            for (int r = 0; r < 8; ++r){
              int bus = t*16+mb+r;
              float vn = v[bus] + (c.f[r]+b4v)*ALPHA*nongen[bus];
              v[bus] = vn; accv2 += vn*vn;
            }
          }
        } else {
          if (n0 < LATENT){
#pragma unroll
            for (int r = 0; r < 8; ++r){ int bus = t*16+mb+r; m[bus*LATENT+n0] += (c.f[r]+b4m)*ALPHA; }
          }
        }
      }
      __syncthreads();
    }
  }
  float bs = block_sum(accv2, shred);
  if (threadIdx.x == 0) atomicAdd(sumv2, bs);
}

// ---- physics kernels (fp32) ------------------------------------------------

__global__ void k_delta_pref(const float* __restrict__ lines, const float* __restrict__ v,
                             const float* __restrict__ theta,
                             float* __restrict__ dpref, float* __restrict__ vpref){
  int i = blockIdx.x*blockDim.x + threadIdx.x;
  if (i >= NBUS) return;
  int s = (int)lines[i*7] - 1, d = (int)lines[i*7+1] - 1;
  dpref[i] = theta[s] - theta[d];
  vpref[i] = v[s] - v[d];
}

__global__ __launch_bounds__(256) void k_global_comp_edges(
    const float* __restrict__ lines, const float* __restrict__ v, const float* __restrict__ theta,
    const float* __restrict__ ypref, const float* __restrict__ vpref,
    float* __restrict__ aggr_from, float* __restrict__ aggr_to, float* __restrict__ scal)
{
  __shared__ float shred[8];
  int e = blockIdx.x*blockDim.x + threadIdx.x;
  float msg = 0.f;
  if (e < NEDGE){
    int s = (int)lines[e*7] - 1, d = (int)lines[e*7+1] - 1;
    float r = lines[e*7+2], x = lines[e*7+3], b = lines[e*7+4], tau = lines[e*7+5];
    float y = 1.f / sqrtf(r*r + x*x);
    float shift = atan2f(r, x);
    float ths = theta[s], thd = theta[d], vs = v[s], vd = v[d];
    float delta = ths - thd;
    msg = fabsf(vs*vd*y/tau*(sinf(ths-thd-delta-shift) + sinf(thd-ths-delta+shift))
              + vs/(tau*tau)*y*sinf(delta) + vd*vd*y*sinf(delta));
    float ys = ypref[s], yd = ypref[d];
    float dvs = vpref[s], dvd = vpref[d];
    float vst = vs/tau;
    float mf = -vs*vd*ys/tau*cosf(ths-thd-dvs-shift) + vst*vst*(ys*cosf(dvs) - 0.5f*b);
    float mt = -vd*vs*yd/tau*cosf(thd-ths-dvd-shift) + vd*vd*(yd*sinf(dvd) - 0.5f*b);
    atomicAdd(&aggr_from[d], mf);
    atomicAdd(&aggr_to[s],  mt);
  }
  float bs = block_sum(msg, shred);
  if (threadIdx.x == 0) atomicAdd(&scal[6], bs);   // p_joule
}

__global__ void k_gen_comp(const float* __restrict__ gens, const float* __restrict__ scal,
                           float* __restrict__ zpg){
  int g = blockIdx.x*blockDim.x + threadIdx.x;
  if (g >= NGEN) return;
  float p_global = scal[5]*scal[0] + scal[1] + scal[6];   // sumv2*sumGs + sumPd + p_joule
  float set_s = scal[2], min_s = scal[3], max_s = scal[4];
  float lam = (p_global < set_s) ? (p_global - min_s) / (2.f*(set_s - min_s))
                                 : (p_global - 2.f*set_s + max_s) / (2.f*(max_s - set_s));
  float sg = gens[g*6+3], mn = gens[g*6+4], mx = gens[g*6+5];
  float pg = (lam < 0.5f) ? (mn + 2.f*(sg - mn)*lam) : (2.f*sg - mx + 2.f*(mx - sg)*lam);
  int gb = (int)gens[g*6] - 1;
  zpg[gb] = pg;
}

__global__ void k_bus_start(const float* __restrict__ buses, const float* __restrict__ v,
                            const float* __restrict__ aggr_from, const float* __restrict__ aggr_to,
                            const float* __restrict__ zpg,
                            float* __restrict__ dp, float* __restrict__ dq){
  int i = blockIdx.x*blockDim.x + threadIdx.x;
  if (i >= NBUS) return;
  float vv = v[i], v2 = vv*vv;
  float Pd = buses[i*6+1], Qd = buses[i*6+2], Gs = buses[i*6+3], Bs = buses[i*6+4];
  float qg = Qd - Bs*v2 - aggr_from[i] - aggr_to[i];
  dp[i] = -Pd - Gs*v2 + zpg[i];
  dq[i] = qg - Qd - Bs*v2;
}

__global__ __launch_bounds__(256) void k_local_edges(
    const float* __restrict__ lines, const float* __restrict__ v, const float* __restrict__ theta,
    const float* __restrict__ ypref, const float* __restrict__ dpref,
    float* __restrict__ dp, float* __restrict__ dq)
{
  int e = blockIdx.x*blockDim.x + threadIdx.x;
  if (e >= NEDGE) return;
  int s = (int)lines[e*7] - 1, d = (int)lines[e*7+1] - 1;
  float r = lines[e*7+2], x = lines[e*7+3], b = lines[e*7+4], tau = lines[e*7+5];
  float shift = atan2f(r, x);
  float ths = theta[s], thd = theta[d], vs = v[s], vd = v[d];
  float ys = ypref[s], yd = ypref[d];
  float ds_ = dpref[s], dd_ = dpref[d];
  float vst = vs/tau;
  float pf = vs*vd*ys/tau*sinf(ths-thd-ds_-shift) + vst*vst*ys*sinf(ds_);
  float pt = vd*vs*yd/tau*sinf(thd-ths-dd_-shift) + vd*vd*yd*sinf(dd_);
  atomicAdd(&dp[d], pf);
  atomicAdd(&dp[s], pt);
  float qf = -vs*vd*ys/tau*cosf(ths-thd-ds_-shift) + vst*vst*(ys*cosf(ds_) - 0.5f*b);
  float qt = -vd*vs*yd/tau*sinf(thd-ths-dd_-shift) + vd*vd*(yd*cosf(dd_) - 0.5f*b);
  atomicAdd(&dq[d], qf);
  atomicAdd(&dq[s], qt);
}

__global__ __launch_bounds__(256) void k_loss(const float* __restrict__ dp, const float* __restrict__ dq,
                                              float* __restrict__ scal, float scale){
  __shared__ float shred[8];
  int i = blockIdx.x*blockDim.x + threadIdx.x;
  float val = 0.f;
  if (i < NBUS) val = dp[i]*dp[i] + dq[i]*dq[i];
  float bs = block_sum(val, shred);
  if (threadIdx.x == 0) atomicAdd(&scal[7], bs*scale);
}

// ---- init / zero / finalize ------------------------------------------------

__global__ void k_zero_scal(float* scal){ if (threadIdx.x < 16) scal[threadIdx.x] = 0.f; }

__global__ __launch_bounds__(256) void k_init_bus(const float* __restrict__ buses, const float* __restrict__ lines,
    float* v, float* theta, float* m, float* nongen, float* ypref, float* scal){
  __shared__ float shred[8];
  int i = blockIdx.x*blockDim.x + threadIdx.x;
  float gs = 0.f, pd = 0.f;
  if (i < NBUS){
    v[i] = 1.f; theta[i] = 0.f; nongen[i] = 1.f;
    for (int j = 0; j < LATENT; ++j) m[i*LATENT+j] = 0.f;
    float r = lines[i*7+2], x = lines[i*7+3];
    ypref[i] = 1.f / sqrtf(r*r + x*x);
    gs = buses[i*6+3]; pd = buses[i*6+1];
  }
  float t = block_sum(gs, shred); if (threadIdx.x == 0) atomicAdd(&scal[0], t);
        t = block_sum(pd, shred); if (threadIdx.x == 0) atomicAdd(&scal[1], t);
}

__global__ __launch_bounds__(256) void k_init_gen(const float* __restrict__ gens,
    float* v, float* nongen, float* scal){
  __shared__ float shred[8];
  int g = blockIdx.x*blockDim.x + threadIdx.x;
  float ss = 0.f, mn = 0.f, mx = 0.f;
  if (g < NGEN){
    int gb = (int)gens[g*6] - 1;
    v[gb] = gens[g*6+2];
    nongen[gb] = 0.f;
    ss = gens[g*6+3]; mn = gens[g*6+4]; mx = gens[g*6+5];
  }
  float t = block_sum(ss, shred); if (threadIdx.x == 0) atomicAdd(&scal[2], t);
        t = block_sum(mn, shred); if (threadIdx.x == 0) atomicAdd(&scal[3], t);
        t = block_sum(mx, shred); if (threadIdx.x == 0) atomicAdd(&scal[4], t);
}

__global__ void k_init_dpq(const float* __restrict__ buses, const float* __restrict__ v,
                           float* dp, float* dq){
  int i = blockIdx.x*blockDim.x + threadIdx.x;
  if (i >= NBUS) return;
  float vv = v[i], v2 = vv*vv;
  dp[i] = -buses[i*6+1] - buses[i*6+3]*v2;
  dq[i] =  buses[i*6+5] - buses[i*6+2] - buses[i*6+4]*v2;
}

__global__ void k_init_genpg(const float* __restrict__ gens, float* dp){
  int g = blockIdx.x*blockDim.x + threadIdx.x;
  if (g >= NGEN) return;
  int gb = (int)gens[g*6] - 1;
  atomicAdd(&dp[gb], gens[g*6+1]);
}

__global__ void k_iter_zero(float* phi, float* af, float* at_, float* zpg, float* scal){
  int i = blockIdx.x*blockDim.x + threadIdx.x;
  if (i < NBUS*LATENT) phi[i] = 0.f;
  if (i < NBUS){ af[i] = 0.f; at_[i] = 0.f; zpg[i] = 0.f; }
  if (i == 0){ scal[5] = 0.f; scal[6] = 0.f; }
}

__global__ void k_final(const float* v, const float* theta, const float* scal, float* out){
  int i = blockIdx.x*blockDim.x + threadIdx.x;
  if (i < NBUS){ out[i] = v[i]; out[NBUS+i] = theta[i]; }
  if (i == 0) out[2*NBUS] = scal[7];
}

// ---- host ------------------------------------------------------------------

extern "C" void kernel_launch(void* const* d_in, const int* in_sizes, int n_in,
                              void* d_out, int out_size, void* d_ws, size_t ws_size,
                              hipStream_t stream)
{
  (void)out_size; (void)ws_size;
  const float* buses = (const float*)d_in[0];
  const float* lines = (const float*)d_in[1];
  const float* gens  = (const float*)d_in[2];

  // Param table: my groups 0=phi_from,1=L_theta,2=L_v,3=L_m; slots W1,B1,W2,B2,W4,B4
  const float* P[4][KITER][6];
  static const int gmap_sorted[4] = {3,1,2,0}, gmap_ins[4] = {0,1,2,3};
  static const int lmap_sorted[6] = {0,3,1,4,2,5}, lmap_ins[6] = {0,1,2,3,4,5};
  if (n_in >= 200){
    bool g_ins = (in_sizes[3] == 300);   // phi_from.W1 first => insertion-order groups
    bool l_ins = (in_sizes[4] == 20);    // b1 second => insertion-order leaves
    const int* gmap = g_ins ? gmap_ins : gmap_sorted;
    const int* lmap = l_ins ? lmap_ins : lmap_sorted;
    for (int g = 0; g < 4; ++g)
      for (int k = 0; k < KITER; ++k)
        for (int s = 0; s < 6; ++s)
          P[g][k][s] = (const float*)d_in[3 + gmap[g]*60 + k*6 + lmap[s]];
  } else {
    // single flat params buffer, jax-sorted order
    const float* p = (const float*)d_in[3];
    static const int SZ[4][6] = { {480,400,200,20,20,10},{480,400,20,20,20,1},
                                  {480,400,20,20,20,1},{300,400,200,20,20,10} };
    const float* F[4][KITER][6];
    long off = 0;
    for (int fg = 0; fg < 4; ++fg)
      for (int k = 0; k < KITER; ++k)
        for (int fl = 0; fl < 6; ++fl){ F[fg][k][fl] = p + off; off += SZ[fg][fl]; }
    for (int g = 0; g < 4; ++g)
      for (int k = 0; k < KITER; ++k)
        for (int s = 0; s < 6; ++s)
          P[g][k][s] = F[gmap_sorted[g]][k][lmap_sorted[s]];
  }

  float* ws     = (float*)d_ws;
  float* m      = ws; ws += NBUS*LATENT;
  float* phi    = ws; ws += NBUS*LATENT;
  float* v      = ws; ws += NBUS;
  float* theta  = ws; ws += NBUS;
  float* dp     = ws; ws += NBUS;
  float* dq     = ws; ws += NBUS;
  float* af     = ws; ws += NBUS;
  float* at_    = ws; ws += NBUS;
  float* zpg    = ws; ws += NBUS;
  float* ypref  = ws; ws += NBUS;
  float* dpref  = ws; ws += NBUS;
  float* vpref  = ws; ws += NBUS;
  float* nongen = ws; ws += NBUS;
  float* scal   = ws; ws += 16;

  int gbus  = (NBUS + 255) / 256;
  int ggen  = (NGEN + 255) / 256;
  int gedge = (NEDGE + 255) / 256;
  int gz    = (NBUS*LATENT + 255) / 256;

  k_zero_scal<<<1, 64, 0, stream>>>(scal);
  k_init_bus <<<gbus, 256, 0, stream>>>(buses, lines, v, theta, m, nongen, ypref, scal);
  k_init_gen <<<ggen, 256, 0, stream>>>(gens, v, nongen, scal);
  k_init_dpq <<<gbus, 256, 0, stream>>>(buses, v, dp, dq);
  k_init_genpg<<<ggen, 256, 0, stream>>>(gens, dp);

  for (int k = 0; k < KITER; ++k){
    float scale = powf(0.9f, (float)(KITER - k)) / (float)NBUS;
    k_iter_zero<<<gz, 256, 0, stream>>>(phi, af, at_, zpg, scal);
    k_edge_mlp<<<640, 256, 0, stream>>>(lines, m, phi,
        P[0][k][0], P[0][k][1], P[0][k][2], P[0][k][3], P[0][k][4], P[0][k][5]);
    MlpW wt = { P[1][k][0], P[1][k][1], P[1][k][2], P[1][k][3], P[1][k][4], P[1][k][5] };
    MlpW wv = { P[2][k][0], P[2][k][1], P[2][k][2], P[2][k][3], P[2][k][4], P[2][k][5] };
    MlpW wm = { P[3][k][0], P[3][k][1], P[3][k][2], P[3][k][3], P[3][k][4], P[3][k][5] };
    k_bus_mlp<<<200, 256, 0, stream>>>(v, theta, dp, dq, m, phi, nongen, &scal[5], wt, wv, wm);
    k_delta_pref<<<gbus, 256, 0, stream>>>(lines, v, theta, dpref, vpref);
    k_global_comp_edges<<<gedge, 256, 0, stream>>>(lines, v, theta, ypref, vpref, af, at_, scal);
    k_gen_comp<<<ggen, 256, 0, stream>>>(gens, scal, zpg);
    k_bus_start<<<gbus, 256, 0, stream>>>(buses, v, af, at_, zpg, dp, dq);
    k_local_edges<<<gedge, 256, 0, stream>>>(lines, v, theta, ypref, dpref, dp, dq);
    k_loss<<<gbus, 256, 0, stream>>>(dp, dq, scal, scale);
  }
  k_final<<<gbus, 256, 0, stream>>>(v, theta, scal, (float*)d_out);
}
